// AttentionBlock_45354854646331
// MI455X (gfx1250) — matrix operand order repaired and checked by reading, not compiled, not run on hardware
//
#include <hip/hip_runtime.h>

// ---------------------------------------------------------------------------
// AttentionBlock for MI455X (gfx1250, wave32, WMMA 16x16x32 f16 -> f32 acc)
// B=8, H=W=32 (seq=1024), C=512, heads=8, head_dim=64, groups=32
// Tile staging: GLOBAL_LOAD_ASYNC_TO_LDS_B128, double-buffered ping-pong
// (copy tile t+1 overlaps WMMA on tile t; one wait+barrier per K-step).
// ---------------------------------------------------------------------------

typedef __attribute__((ext_vector_type(16))) _Float16 v16h;
typedef __attribute__((ext_vector_type(8)))  _Float16 v8h;
typedef __attribute__((ext_vector_type(4)))  _Float16 v4h;
typedef __attribute__((ext_vector_type(8)))  float    v8f;
typedef __attribute__((ext_vector_type(4)))  float    v4f;

#define BATCH   8
#define SEQ     1024
#define CH      512
#define NHEAD   8
#define HDIM    64
#define NGROUP  32
#define MTOT    (BATCH * SEQ)          // 8192 rows
#define QKN     (3 * CH)               // 1536

// -------------------- CDNA5 async copy (global -> LDS, 16B per lane) --------

static __device__ __forceinline__ void async_copy_b128(const _Float16* gsrc,
                                                       const _Float16* ldst) {
    // Low 32 bits of a flat shared-space address are the LDS byte address.
    const unsigned lds_addr = (unsigned)(unsigned long long)ldst;
    const unsigned long long gaddr = (unsigned long long)gsrc;
    asm volatile("global_load_async_to_lds_b128 %0, %1, off"
                 :: "v"(lds_addr), "v"(gaddr)
                 : "memory");
}

static __device__ __forceinline__ void wait_async_all() {
    asm volatile("s_wait_asynccnt 0x0" ::: "memory");
}

// -------------------- WMMA fragment helpers (per CDNA5 ISA layouts) ---------

static __device__ __forceinline__ v8f wmma_f16(v16h a, v16h b, v8f c) {
    return __builtin_amdgcn_wmma_f32_16x16x32_f16(
        false, a, false, b, (short)0, c, false, false);
}

// A-matrix 16x32 f16, logical A[m][k] = base[m*ld + k]. Two 16B loads/lane.
static __device__ __forceinline__ v16h load_frag_a(const _Float16* base, int ld) {
    const int lane = threadIdx.x & 31;
    const int m    = lane & 15;
    const int khi  = (lane >> 4) ? 8 : 0;
    union { v16h v; v8h p[2]; } u;
    u.p[0] = *(const v8h*)(base + m * ld + khi);
    u.p[1] = *(const v8h*)(base + m * ld + 16 + khi);
    return u.v;
}

// B-matrix 32x16 f16 from an n-major buffer: B[k][n] = base[n*ld + k].
static __device__ __forceinline__ v16h load_frag_bT(const _Float16* base, int ld) {
    const int lane = threadIdx.x & 31;
    const int n    = lane & 15;
    const int kb   = (lane >> 4) ? 8 : 0;
    union { v16h v; v8h p[2]; } u;
    u.p[0] = *(const v8h*)(base + n * ld + kb);
    u.p[1] = *(const v8h*)(base + n * ld + kb + 16);
    return u.v;
}

// C/D 16x16 f32 store: VGPR i holds row (i + 8*laneHi), col = lane&15.
static __device__ __forceinline__ void store_frag_f32(float* base, int ld, v8f c) {
    const int lane = threadIdx.x & 31;
    const int n    = lane & 15;
    const int mb   = (lane >> 4) ? 8 : 0;
#pragma unroll
    for (int i = 0; i < 8; ++i)
        base[(mb + i) * ld + n] = c[i];
}

// -------------------- Kernel 1: GroupNorm -> f16 (vectorized) ---------------
__global__ __launch_bounds__(256)
void gn_to_f16_kernel(const float* __restrict__ x,
                      const float* __restrict__ gs,
                      const float* __restrict__ gb,
                      _Float16* __restrict__ xn) {
    __shared__ float ssum[256];
    __shared__ float ssq[256];
    const int b = blockIdx.x >> 5;
    const int g = blockIdx.x & 31;
    const int t = threadIdx.x;

    float sum = 0.f, sq = 0.f;
    for (int v = t; v < SEQ * 4; v += 256) {       // 4 x float4 per spatial row
        const int s = v >> 2, c4 = v & 3;
        const v4f d = *(const v4f*)&x[(b * SEQ + s) * CH + g * 16 + c4 * 4];
#pragma unroll
        for (int j = 0; j < 4; ++j) { sum += d[j]; sq += d[j] * d[j]; }
    }
    ssum[t] = sum; ssq[t] = sq;
    __syncthreads();
#pragma unroll
    for (int ofs = 128; ofs > 0; ofs >>= 1) {
        if (t < ofs) { ssum[t] += ssum[t + ofs]; ssq[t] += ssq[t + ofs]; }
        __syncthreads();
    }
    const float inv_n = 1.0f / (float)(SEQ * 16);
    const float mu  = ssum[0] * inv_n;
    const float var = ssq[0] * inv_n - mu * mu;
    const float rs  = rsqrtf(var + 1e-6f);

    v4f sc[4], bi[4];
#pragma unroll
    for (int c4 = 0; c4 < 4; ++c4) {
        sc[c4] = *(const v4f*)&gs[g * 16 + c4 * 4];
        bi[c4] = *(const v4f*)&gb[g * 16 + c4 * 4];
    }
    for (int v = t; v < SEQ * 4; v += 256) {
        const int s = v >> 2, c4 = v & 3;
        const v4f d = *(const v4f*)&x[(b * SEQ + s) * CH + g * 16 + c4 * 4];
        v4h o;
#pragma unroll
        for (int j = 0; j < 4; ++j)
            o[j] = (_Float16)((d[j] - mu) * rs * sc[c4][j] + bi[c4][j]);
        *(v4h*)&xn[(b * SEQ + s) * CH + g * 16 + c4 * 4] = o;
    }
}

// -------------------- Kernel 2: f32 -> f16 transposed convert ---------------
// in: K x N row-major f32; out: N x K row-major f16.  grid = (N/256, K).
__global__ __launch_bounds__(256)
void cvt_f16_t_kernel(const float* __restrict__ in, _Float16* __restrict__ out,
                      int K, int N) {
    const int n = blockIdx.x * 256 + threadIdx.x;
    const int k = blockIdx.y;
    out[n * K + k] = (_Float16)in[k * N + n];
}

// -------------------- Kernel 3: QKV GEMM (8192x1536x512) --------------------
// Block tile 128x64, K-step 64, double-buffered async staging.
// 8 waves each own 32x32 -> 8 WMMAs per K-step per wave.
__global__ __launch_bounds__(256)
void qkv_gemm_kernel(const _Float16* __restrict__ xn,
                     const _Float16* __restrict__ wqT,   // [1536][512] n-major
                     const float*    __restrict__ bq,
                     _Float16* __restrict__ qh,          // [bh][s][d]
                     _Float16* __restrict__ kh,          // [bh][s][d]
                     _Float16* __restrict__ vh) {        // [bh][d][s] d-major
    __shared__ alignas(16) _Float16 lds_a[2][128 * 64];  // 2 x 16 KB
    __shared__ alignas(16) _Float16 lds_b[2][64 * 64];   // 2 x  8 KB

    const int m0 = blockIdx.y * 128;
    const int n0 = blockIdx.x * 64;
    const int w  = threadIdx.x >> 5;
    const int wm = w & 3;
    const int wn = w >> 2;

    auto stage = [&](int kt, int buf) {
        const int idx4 = threadIdx.x * 4;
#pragma unroll
        for (int u = 0; u < 4; ++u) {               // A tile 128x64
            const int v = idx4 + u, r = v >> 3, c8 = v & 7;
            async_copy_b128(&xn[(m0 + r) * CH + kt + c8 * 8],
                            &lds_a[buf][r * 64 + c8 * 8]);
        }
#pragma unroll
        for (int u = 0; u < 2; ++u) {               // B tile 64(n) x 64(k)
            const int v = threadIdx.x * 2 + u, n = v >> 3, k8 = v & 7;
            async_copy_b128(&wqT[(n0 + n) * CH + kt + k8 * 8],
                            &lds_b[buf][n * 64 + k8 * 8]);
        }
    };

    v8f acc[2][2] = {};
    stage(0, 0);
    for (int it = 0; it < CH / 64; ++it) {
        const int cur = it & 1;
        wait_async_all();
        __syncthreads();
        if (it + 1 < CH / 64) stage((it + 1) * 64, cur ^ 1);

        const _Float16* ab = lds_a[cur];
        const _Float16* bb = lds_b[cur];
#pragma unroll
        for (int kc = 0; kc < 2; ++kc) {
            v16h af0 = load_frag_a (ab + (wm * 32) * 64 + kc * 32, 64);
            v16h af1 = load_frag_a (ab + (wm * 32 + 16) * 64 + kc * 32, 64);
            v16h bf0 = load_frag_bT(bb + (wn * 32) * 64 + kc * 32, 64);
            v16h bf1 = load_frag_bT(bb + (wn * 32 + 16) * 64 + kc * 32, 64);
            acc[0][0] = wmma_f16(af0, bf0, acc[0][0]);
            acc[0][1] = wmma_f16(af0, bf1, acc[0][1]);
            acc[1][0] = wmma_f16(af1, bf0, acc[1][0]);
            acc[1][1] = wmma_f16(af1, bf1, acc[1][1]);
        }
    }

    // Epilogue: sect / head / batch are uniform for the whole block.
    const int lane = threadIdx.x & 31;
    const int nl   = lane & 15;
    const int mb   = (lane >> 4) ? 8 : 0;
    const int sect = n0 >> 9;                 // 0=q 1=k 2=v (uniform)
    const int hh   = (n0 & 511) >> 6;         // head (uniform)
    const int b    = m0 >> 10;                // batch (uniform)
    const int bh   = b * NHEAD + hh;
    const int s0   = (m0 & 1023) + wm * 32;
    const float scale = (sect < 2) ? 0.125f : 1.0f;
    _Float16* __restrict__ qk = (sect == 0) ? qh : kh;

#pragma unroll
    for (int ni = 0; ni < 2; ++ni) {
        const int d    = wn * 32 + ni * 16 + nl;
        const float bias = bq[n0 + d];
#pragma unroll
        for (int mi = 0; mi < 2; ++mi)
#pragma unroll
            for (int i = 0; i < 8; ++i) {
                const int s = s0 + mi * 16 + mb + i;
                const float v = (acc[mi][ni][i] + bias) * scale;
                if (sect < 2)
                    qk[(bh * SEQ + s) * HDIM + d] = (_Float16)v;
                else
                    vh[(bh * HDIM + d) * SEQ + s] = (_Float16)v;
            }
    }
}

// -------------------- Kernel 4: flash attention -----------------------------
// One block per (b, head, 128-query tile); 8 waves x 16 query rows.
// K/V tiles double-buffered via async copies (32 keys per step).
__global__ __launch_bounds__(256)
void flash_attn_kernel(const _Float16* __restrict__ qh,
                       const _Float16* __restrict__ kh,   // [bh][s][d]
                       const _Float16* __restrict__ vh,   // [bh][d][s]
                       _Float16* __restrict__ attn) {
    __shared__ alignas(16) _Float16 Kt[2][32 * 64];       // 2 x 4 KB [key][d]
    __shared__ alignas(16) _Float16 Vt[2][64 * 32];       // 2 x 4 KB [d][key]
    __shared__ alignas(16) float    Sl[8][16 * 32];       // 16 KB per-wave S
    __shared__ alignas(16) _Float16 Pl[8][16 * 32];       // 8 KB  per-wave P
    __shared__ float rmax[8][16];
    __shared__ float rsum[8][16];
    __shared__ float resc[8][16];

    const int bh = blockIdx.x >> 3;
    const int qt = blockIdx.x & 7;
    const int w    = threadIdx.x >> 5;
    const int lane = threadIdx.x & 31;
    const int nl   = lane & 15;
    const int mb   = (lane >> 4) ? 8 : 0;
    const int base = bh * SEQ * HDIM;

    auto stageKV = [&](int kt, int buf) {
        const int r  = threadIdx.x >> 3, c8 = threadIdx.x & 7;
        async_copy_b128(&kh[base + (kt + r) * HDIM + c8 * 8],
                        &Kt[buf][r * 64 + c8 * 8]);
        const int d  = threadIdx.x >> 2, s8 = threadIdx.x & 3;
        async_copy_b128(&vh[base + d * SEQ + kt + s8 * 8],
                        &Vt[buf][d * 32 + s8 * 8]);
    };

    // Q fragments (16 rows x 64 d) held in registers for the whole loop
    const _Float16* qp = qh + base + (qt * 128 + w * 16) * HDIM;
    v16h aq0 = load_frag_a(qp, HDIM);
    v16h aq1 = load_frag_a(qp + 32, HDIM);

    if (lane < 16) {
        rmax[w][lane] = -__builtin_inff();
        rsum[w][lane] = 0.0f;
    }
    v8f o[4] = {};

    stageKV(0, 0);
    for (int it = 0; it < SEQ / 32; ++it) {
        const int cur = it & 1;
        wait_async_all();
        __syncthreads();
        if (it + 1 < SEQ / 32) stageKV((it + 1) * 32, cur ^ 1);

        const _Float16* kb = Kt[cur];
        const _Float16* vb = Vt[cur];

        // S (16x32) = Q (16x64) @ K^T (64x32)
#pragma unroll
        for (int ni = 0; ni < 2; ++ni) {
            v8f s = {};
            v16h bk0 = load_frag_bT(kb + (ni * 16) * 64 + 0, 64);
            v16h bk1 = load_frag_bT(kb + (ni * 16) * 64 + 32, 64);
            s = wmma_f16(aq0, bk0, s);
            s = wmma_f16(aq1, bk1, s);
            store_frag_f32(&Sl[w][ni * 16], 32, s);
        }

        // Online softmax: lanes 0..15 each own one of the wave's 16 rows.
        if (lane < 16) {
            const float mold = rmax[w][lane];
            float mloc = mold;
            const float* srow = &Sl[w][lane * 32];
#pragma unroll
            for (int j = 0; j < 32; ++j) mloc = fmaxf(mloc, srow[j]);
            const float rf = __expf(mold - mloc);     // 0 on first tile
            float sum = 0.0f;
            _Float16* prow = &Pl[w][lane * 32];
#pragma unroll
            for (int j = 0; j < 32; ++j) {
                const float p = __expf(srow[j] - mloc);
                prow[j] = (_Float16)p;
                sum += p;
            }
            rmax[w][lane] = mloc;
            rsum[w][lane] = rsum[w][lane] * rf + sum;
            resc[w][lane] = rf;
        }

        // O = O*rescale + P @ V
#pragma unroll
        for (int nc = 0; nc < 4; ++nc)
#pragma unroll
            for (int i = 0; i < 8; ++i)
                o[nc][i] *= resc[w][mb + i];

        v16h ap = load_frag_a(&Pl[w][0], 32);
#pragma unroll
        for (int nc = 0; nc < 4; ++nc) {
            v16h bv = load_frag_bT(vb + (nc * 16) * 32, 32);
            o[nc] = wmma_f16(ap, bv, o[nc]);
        }
    }

    // Normalize (reciprocal once per row) and store to [b, s, h*64+d] f16.
    const int b = bh >> 3, h = bh & 7;
    float rinv[8];
#pragma unroll
    for (int i = 0; i < 8; ++i) rinv[i] = 1.0f / rsum[w][mb + i];
#pragma unroll
    for (int nc = 0; nc < 4; ++nc)
#pragma unroll
        for (int i = 0; i < 8; ++i) {
            const int srow = qt * 128 + w * 16 + mb + i;
            attn[(b * SEQ + srow) * CH + h * HDIM + nc * 16 + nl] =
                (_Float16)(o[nc][i] * rinv[i]);
        }
}

// -------------------- Kernel 5: out projection + residual -------------------
__global__ __launch_bounds__(256)
void out_proj_kernel(const _Float16* __restrict__ attn,
                     const _Float16* __restrict__ woT,   // [512][512] n-major
                     const float*    __restrict__ bo,
                     const float*    __restrict__ x,
                     float* __restrict__ out) {
    __shared__ alignas(16) _Float16 lds_a[2][128 * 64];
    __shared__ alignas(16) _Float16 lds_b[2][64 * 64];

    const int m0 = blockIdx.y * 128;
    const int n0 = blockIdx.x * 64;
    const int w  = threadIdx.x >> 5;
    const int wm = w & 3;
    const int wn = w >> 2;

    auto stage = [&](int kt, int buf) {
        const int idx4 = threadIdx.x * 4;
#pragma unroll
        for (int u = 0; u < 4; ++u) {
            const int v = idx4 + u, r = v >> 3, c8 = v & 7;
            async_copy_b128(&attn[(m0 + r) * CH + kt + c8 * 8],
                            &lds_a[buf][r * 64 + c8 * 8]);
        }
#pragma unroll
        for (int u = 0; u < 2; ++u) {
            const int v = threadIdx.x * 2 + u, n = v >> 3, k8 = v & 7;
            async_copy_b128(&woT[(n0 + n) * CH + kt + k8 * 8],
                            &lds_b[buf][n * 64 + k8 * 8]);
        }
    };

    v8f acc[2][2] = {};
    stage(0, 0);
    for (int it = 0; it < CH / 64; ++it) {
        const int cur = it & 1;
        wait_async_all();
        __syncthreads();
        if (it + 1 < CH / 64) stage((it + 1) * 64, cur ^ 1);

        const _Float16* ab = lds_a[cur];
        const _Float16* bb = lds_b[cur];
#pragma unroll
        for (int kc = 0; kc < 2; ++kc) {
            v16h af0 = load_frag_a (ab + (wm * 32) * 64 + kc * 32, 64);
            v16h af1 = load_frag_a (ab + (wm * 32 + 16) * 64 + kc * 32, 64);
            v16h bf0 = load_frag_bT(bb + (wn * 32) * 64 + kc * 32, 64);
            v16h bf1 = load_frag_bT(bb + (wn * 32 + 16) * 64 + kc * 32, 64);
            acc[0][0] = wmma_f16(af0, bf0, acc[0][0]);
            acc[0][1] = wmma_f16(af0, bf1, acc[0][1]);
            acc[1][0] = wmma_f16(af1, bf0, acc[1][0]);
            acc[1][1] = wmma_f16(af1, bf1, acc[1][1]);
        }
    }

    const int lane = threadIdx.x & 31;
    const int nl = lane & 15;
    const int mb = (lane >> 4) ? 8 : 0;
#pragma unroll
    for (int ni = 0; ni < 2; ++ni) {
        const int n = n0 + wn * 32 + ni * 16 + nl;
        const float bias = bo[n];
#pragma unroll
        for (int mi = 0; mi < 2; ++mi)
#pragma unroll
            for (int i = 0; i < 8; ++i) {
                const int m = m0 + wm * 32 + mi * 16 + mb + i;
                out[m * CH + n] = acc[mi][ni][i] + bias + x[m * CH + n];
            }
    }
}

// -------------------- host launch -------------------------------------------
extern "C" void kernel_launch(void* const* d_in, const int* in_sizes, int n_in,
                              void* d_out, int out_size, void* d_ws, size_t ws_size,
                              hipStream_t stream) {
    const float* x    = (const float*)d_in[0];
    const float* gs   = (const float*)d_in[1];
    const float* gb   = (const float*)d_in[2];
    const float* wqkv = (const float*)d_in[3];
    const float* bqkv = (const float*)d_in[4];
    const float* wout = (const float*)d_in[5];
    const float* bout = (const float*)d_in[6];
    float* out = (float*)d_out;

    _Float16* ws = (_Float16*)d_ws;
    size_t o = 0;
    _Float16* xn   = ws + o; o += (size_t)MTOT * CH;                  // 4 Mi
    _Float16* wqT  = ws + o; o += (size_t)QKN * CH;                   // 768 Ki
    _Float16* woT  = ws + o; o += (size_t)CH * CH;                    // 256 Ki
    _Float16* qh   = ws + o; o += (size_t)BATCH * NHEAD * SEQ * HDIM; // 4 Mi
    _Float16* kh   = ws + o; o += (size_t)BATCH * NHEAD * SEQ * HDIM; // 4 Mi
    _Float16* vh   = ws + o; o += (size_t)BATCH * NHEAD * SEQ * HDIM; // 4 Mi
    _Float16* attn = ws + o;                                          // 4 Mi

    // 1) GroupNorm -> f16
    gn_to_f16_kernel<<<BATCH * NGROUP, 256, 0, stream>>>(x, gs, gb, xn);

    // 2) weight conversion (transposed, n-major, for contiguous B fragments)
    cvt_f16_t_kernel<<<dim3(QKN / 256, CH), 256, 0, stream>>>(wqkv, wqT, CH, QKN);
    cvt_f16_t_kernel<<<dim3(CH / 256, CH), 256, 0, stream>>>(wout, woT, CH, CH);

    // 3) QKV GEMM -> head-major q,k (pre-scaled) and d-major v
    qkv_gemm_kernel<<<dim3(QKN / 64, MTOT / 128), 256, 0, stream>>>(
        xn, wqT, bqkv, qh, kh, vh);

    // 4) flash attention
    flash_attn_kernel<<<BATCH * NHEAD * (SEQ / 128), 256, 0, stream>>>(
        qh, kh, vh, attn);

    // 5) out projection + bias + residual
    out_proj_kernel<<<dim3(CH / 64, MTOT / 128), 256, 0, stream>>>(
        attn, woT, bout, x, out);
}
